// NeuroSAT_33414845563193
// MI455X (gfx1250) — matrix operand order repaired and checked by reading, not compiled, not run on hardware
//
#include <hip/hip_runtime.h>
#include <hip/hip_bf16.h>
#include <math.h>

typedef __attribute__((ext_vector_type(16))) _Float16 v16h;
typedef __attribute__((ext_vector_type(8)))  float    v8f;
typedef int v4i __attribute__((vector_size(16)));   // matches async-LDS builtin param type

#define EMB       128
#define NLITS     4096
#define NCLAUSES  8192
#define NCELLS    32768
#define NPROB     16
#define N_ITERS   8
#define NT        4          // N-tiles (of 16) per wave -> 16x64 output per wave
#define ASTRIDE   36         // padded LDS row stride (floats) to spread banks

// --- CDNA5 async global->LDS path (guarded; falls back to plain staging) ---
#if defined(__gfx1250__) && __has_builtin(__builtin_amdgcn_global_load_async_to_lds_b128)
#define HAS_ASYNC_LDS 1
#else
#define HAS_ASYNC_LDS 0
#endif

#if HAS_ASYNC_LDS
#if __has_builtin(__builtin_amdgcn_s_wait_asynccnt)
#define WAIT_ASYNC(n) __builtin_amdgcn_s_wait_asynccnt(n)
#else
#define WAIT_ASYNC(n) asm volatile("s_wait_asynccnt " #n ::: "memory")
#endif
#else
#define WAIT_ASYNC(n) asm volatile("" ::: "memory")   // compiler fence; LDS is in-order per wave
#endif

// ---------------------------------------------------------------------------
// Weight packing: W is torch-layout [N x Wld] fp32.  We build B = W^T[kOff:kOff+K, :]
// as f16 in WMMA B-fragment order: tile (kt,nt) of 32x16, lane l holds K-range
// kt*32 + 16*(l/16) + e for column nt*16 + (l%16)   (contiguous k order, same as the A fragment).
// dst[((kt*(N/16)+nt)*32 + l)*16 + e]
// ---------------------------------------------------------------------------
__global__ void pack_b_kernel(const float* __restrict__ W, _Float16* __restrict__ dst,
                              int N, int K, int Wld, int kOff) {
    int tid = blockIdx.x * blockDim.x + threadIdx.x;
    if (tid >= N * K) return;
    int e    = tid & 15;
    int l    = (tid >> 4) & 31;
    int tile = tid >> 9;
    int ntn  = N >> 4;
    int nt   = tile % ntn;
    int kt   = tile / ntn;
    int n = nt * 16 + (l & 15);
    int k = kt * 32 + ((l >> 4) << 4) + e;
    dst[tid] = (_Float16)W[(size_t)n * Wld + kOff + k];
}

// ---------------------------------------------------------------------------
// WMMA GEMM: C[M x N] = act( A[M x K](fp32) * Bpacked(f16) + bias )  or  += .
// 256 threads = 8 waves; wave w owns M-tile (bx*8+w) and 4 N-tiles (by*4..+3).
// A tile (16 rows x 32 K, fp32) is double-buffered in LDS via async loads.
// rowXor remaps A's logical row -> physical row (literal flip = XOR 2048).
// ---------------------------------------------------------------------------
__global__ void __launch_bounds__(256)
gemm_wmma_kernel(const float* __restrict__ A, const _Float16* __restrict__ Bp,
                 const float* __restrict__ bias, float* __restrict__ C,
                 int M, int N, int K, int rowXor, int accumulate, int relu) {
    __shared__ float As[8][2][16 * ASTRIDE] __attribute__((aligned(16)));

    int wave = threadIdx.x >> 5;
    int lane = threadIdx.x & 31;
    int mt  = blockIdx.x * 8 + wave;
    int nt0 = blockIdx.y * NT;
    if (mt * 16 >= M) return;               // uniform per wave; EXEC stays all-1s

    int g  = lane >> 4;                     // K-half selector
    int m  = lane & 15;                     // row within tile / col within n-tile
    int n0 = nt0 * 16 + m;
    int ntn = N >> 4;
    int KT  = K >> 5;

    v8f acc[NT];
    if (accumulate) {
#pragma unroll
        for (int j = 0; j < NT; ++j)
#pragma unroll
            for (int r = 0; r < 8; ++r)
                acc[j][r] = C[(size_t)(mt * 16 + r + 8 * g) * N + n0 + 16 * j];
    } else {
#pragma unroll
        for (int j = 0; j < NT; ++j) {
            float bv = bias ? bias[n0 + 16 * j] : 0.0f;
#pragma unroll
            for (int r = 0; r < 8; ++r) acc[j][r] = bv;
        }
    }

    // --- A-tile staging: lane covers rows (lane>>3)+4i, 16B chunk (lane&7) ---
    int srow = lane >> 3;
    int scol = (lane & 7) * 4;
    auto stage = [&](int kt, int buf) {
#pragma unroll
        for (int i = 0; i < 4; ++i) {
            int row = srow + 4 * i;
            const float* gp = A + (size_t)((mt * 16 + row) ^ rowXor) * K + kt * 32 + scol;
            float*       lp = &As[wave][buf][row * ASTRIDE + scol];
#if HAS_ASYNC_LDS
            __builtin_amdgcn_global_load_async_to_lds_b128(
                (__attribute__((address_space(1))) v4i*)(size_t)gp,
                (__attribute__((address_space(3))) v4i*)(unsigned)(size_t)lp,
                0, 0);
#else
            *(float4*)lp = *(const float4*)gp;
#endif
        }
    };

    stage(0, 0);
    for (int kt = 0; kt < KT; ++kt) {
        int cur = kt & 1;
        if (kt + 1 < KT) {
            stage(kt + 1, cur ^ 1);   // prefetch next tile into other buffer
            WAIT_ASYNC(4);            // retire current tile's 4 B128 transfers
        } else {
            WAIT_ASYNC(0);
        }
        // A fragment: lane (m, half g) holds K = 16g+[0..15] (matches packed B)
        const float4* q0 = (const float4*)&As[wave][cur][m * ASTRIDE + 16 * g];
        const float4* q1 = (const float4*)&As[wave][cur][m * ASTRIDE + 16 * g + 8];
        float4 f0 = q0[0], f1 = q0[1], f2 = q1[0], f3 = q1[1];
        v16h a;
        a[0]=(_Float16)f0.x; a[1]=(_Float16)f0.y; a[2]=(_Float16)f0.z; a[3]=(_Float16)f0.w;
        a[4]=(_Float16)f1.x; a[5]=(_Float16)f1.y; a[6]=(_Float16)f1.z; a[7]=(_Float16)f1.w;
        a[8]=(_Float16)f2.x; a[9]=(_Float16)f2.y; a[10]=(_Float16)f2.z; a[11]=(_Float16)f2.w;
        a[12]=(_Float16)f3.x; a[13]=(_Float16)f3.y; a[14]=(_Float16)f3.z; a[15]=(_Float16)f3.w;
#pragma unroll
        for (int j = 0; j < NT; ++j) {
            v16h b = *(const v16h*)(Bp + ((size_t)(kt * ntn + nt0 + j) * 32 + lane) * 16);
            acc[j] = __builtin_amdgcn_wmma_f32_16x16x32_f16(false, a, false, b, (short)0,
                                                            acc[j], false, false);
        }
    }

#pragma unroll
    for (int j = 0; j < NT; ++j)
#pragma unroll
        for (int r = 0; r < 8; ++r) {
            float v = acc[j][r];
            if (relu) v = v > 0.0f ? v : 0.0f;
            C[(size_t)(mt * 16 + r + 8 * g) * N + n0 + 16 * j] = v;
        }
}

// ---------------------------------------------------------------------------
// LayerNorm over EMB=128, one wave per row (4 floats per lane, shfl reduce).
// ---------------------------------------------------------------------------
__global__ void __launch_bounds__(256)
layernorm_kernel(const float* __restrict__ x, const float* __restrict__ gw,
                 const float* __restrict__ bw, float* __restrict__ y, int rows) {
    int warp = threadIdx.x >> 5, lane = threadIdx.x & 31;
    int row = blockIdx.x * 8 + warp;
    if (row >= rows) return;
    const float4 v  = ((const float4*)(x + (size_t)row * EMB))[lane];
    float s = v.x + v.y + v.z + v.w;
    float q = v.x * v.x + v.y * v.y + v.z * v.z + v.w * v.w;
    for (int m = 16; m; m >>= 1) { s += __shfl_xor(s, m, 32); q += __shfl_xor(q, m, 32); }
    float mean = s * (1.0f / EMB);
    float var  = q * (1.0f / EMB) - mean * mean;
    float inv  = rsqrtf(var + 1e-5f);
    const float4 g4 = ((const float4*)gw)[lane];
    const float4 b4 = ((const float4*)bw)[lane];
    float4 o;
    o.x = (v.x - mean) * inv * g4.x + b4.x;
    o.y = (v.y - mean) * inv * g4.y + b4.y;
    o.z = (v.z - mean) * inv * g4.z + b4.z;
    o.w = (v.w - mean) * inv * g4.w + b4.w;
    ((float4*)(y + (size_t)row * EMB))[lane] = o;
}

// ---------------------------------------------------------------------------
// Sparse scatter: dst[dstIdx[e]] += w[e] * src[srcIdx[e]]  (per-emb-element)
// ---------------------------------------------------------------------------
__global__ void scatter_add_kernel(const int* __restrict__ srcIdx, const int* __restrict__ dstIdx,
                                   const float* __restrict__ w, const float* __restrict__ src,
                                   float* __restrict__ dst) {
    int tid = blockIdx.x * blockDim.x + threadIdx.x;
    if (tid >= NCELLS * EMB) return;
    int e = tid >> 7, j = tid & 127;
    atomicAdd(&dst[(size_t)dstIdx[e] * EMB + j], w[e] * src[(size_t)srcIdx[e] * EMB + j]);
}

// LSTM elementwise: gates [rows x 512] in (i,f,g,o) order; updates h,c in place.
__global__ void lstm_cell_kernel(float* __restrict__ h, float* __restrict__ c,
                                 const float* __restrict__ gates, int rows) {
    int tid = blockIdx.x * blockDim.x + threadIdx.x;
    if (tid >= rows * EMB) return;
    int r = tid >> 7, j = tid & 127;
    const float* gr = gates + (size_t)r * 4 * EMB;
    float gi = gr[j], gf = gr[EMB + j], gg = gr[2 * EMB + j], go = gr[3 * EMB + j];
    float si = 1.0f / (1.0f + __expf(-gi));
    float sf = 1.0f / (1.0f + __expf(-gf));
    float so = 1.0f / (1.0f + __expf(-go));
    float c2 = sf * c[tid] + si * tanhf(gg);
    c[tid] = c2;
    h[tid] = so * tanhf(c2);
}

// Final vote layer: 128-dot per row + bias, write votes_sorted + vote buffer.
__global__ void __launch_bounds__(256)
vote_finish_kernel(const float* __restrict__ t2, const float* __restrict__ w3,
                   const float* __restrict__ b3, float* __restrict__ votes_out,
                   float* __restrict__ voteBuf, int it) {
    int warp = threadIdx.x >> 5, lane = threadIdx.x & 31;
    int row = blockIdx.x * 8 + warp;
    if (row >= NLITS) return;
    const float4 v  = ((const float4*)(t2 + (size_t)row * EMB))[lane];
    const float4 wv = ((const float4*)w3)[lane];
    float s = v.x * wv.x + v.y * wv.y + v.z * wv.z + v.w * wv.w;
    for (int m = 16; m; m >>= 1) s += __shfl_xor(s, m, 32);
    if (lane == 0) {
        float vote = s + b3[0];
        voteBuf[row] = vote;
        int col = row < 2048 ? 0 : 1;
        int rv  = row < 2048 ? row : row - 2048;
        int p = rv >> 7, i = rv & 127;
        votes_out[((size_t)(it * NPROB + p) * 128 + i) * 2 + col] = vote;
    }
}

__global__ void vote_mean_kernel(const float* __restrict__ voteBuf, float* __restrict__ out) {
    __shared__ float sh[128];
    int p = blockIdx.x, i = threadIdx.x;
    sh[i] = voteBuf[p * 128 + i] + voteBuf[2048 + p * 128 + i];
    __syncthreads();
    for (int st = 64; st; st >>= 1) { if (i < st) sh[i] += sh[i + st]; __syncthreads(); }
    if (i == 0) out[p] = sh[0] / 256.0f;
}

__global__ void init_state_kernel(float* __restrict__ h, const float* __restrict__ w,
                                  const float* __restrict__ b, int rows) {
    int tid = blockIdx.x * blockDim.x + threadIdx.x;
    if (tid >= rows * EMB) return;
    int j = tid & 127;
    h[tid] = w[j] + b[j];
}

__global__ void zero_kernel(float* __restrict__ p, int n) {
    int tid = blockIdx.x * blockDim.x + threadIdx.x;
    if (tid < n) p[tid] = 0.0f;
}

__global__ void add_vec_kernel(const float* __restrict__ a, const float* __restrict__ b,
                               float* __restrict__ o, int n) {
    int tid = blockIdx.x * blockDim.x + threadIdx.x;
    if (tid < n) o[tid] = a[tid] + b[tid];
}

// ---------------------------------------------------------------------------
static inline void* bump(char*& p, size_t bytes) {
    void* r = (void*)p;
    p += (bytes + 255) & ~(size_t)255;
    return r;
}

extern "C" void kernel_launch(void* const* d_in, const int* in_sizes, int n_in,
                              void* d_out, int out_size, void* d_ws, size_t ws_size,
                              hipStream_t stream) {
    (void)in_sizes; (void)n_in; (void)out_size; (void)ws_size;

    const int*   edges = (const int*)d_in[0];        // [2, NCELLS]: row0=lit, row1=clause
    const float* ew    = (const float*)d_in[1];
    const float* LiW = (const float*)d_in[2];  const float* LiB = (const float*)d_in[3];
    const float* CiW = (const float*)d_in[4];  const float* CiB = (const float*)d_in[5];
    const float* Lm_w1=(const float*)d_in[6];  const float* Lm_b1=(const float*)d_in[7];
    const float* Lm_w2=(const float*)d_in[8];  const float* Lm_b2=(const float*)d_in[9];
    const float* Lm_w3=(const float*)d_in[10]; const float* Lm_b3=(const float*)d_in[11];
    const float* Cm_w1=(const float*)d_in[12]; const float* Cm_b1=(const float*)d_in[13];
    const float* Cm_w2=(const float*)d_in[14]; const float* Cm_b2=(const float*)d_in[15];
    const float* Cm_w3=(const float*)d_in[16]; const float* Cm_b3=(const float*)d_in[17];
    const float* Lv_w1=(const float*)d_in[18]; const float* Lv_b1=(const float*)d_in[19];
    const float* Lv_w2=(const float*)d_in[20]; const float* Lv_b2=(const float*)d_in[21];
    const float* Lv_w3=(const float*)d_in[22]; const float* Lv_b3=(const float*)d_in[23];
    const float* Lu_wih=(const float*)d_in[24]; const float* Lu_whh=(const float*)d_in[25];
    const float* Lu_bih=(const float*)d_in[26]; const float* Lu_bhh=(const float*)d_in[27];
    const float* Cu_wih=(const float*)d_in[28]; const float* Cu_whh=(const float*)d_in[29];
    const float* Cu_bih=(const float*)d_in[30]; const float* Cu_bhh=(const float*)d_in[31];
    const float* ln_l_g=(const float*)d_in[32]; const float* ln_l_b=(const float*)d_in[33];
    const float* ln_c_g=(const float*)d_in[34]; const float* ln_c_b=(const float*)d_in[35];

    const int* e_lit = edges;
    const int* e_cls = edges + NCELLS;

    // ---- workspace carve-out --------------------------------------------
    char* wp = (char*)d_ws;
    float* L_h   = (float*)bump(wp, (size_t)NLITS   * EMB * 4);
    float* L_c   = (float*)bump(wp, (size_t)NLITS   * EMB * 4);
    float* C_h   = (float*)bump(wp, (size_t)NCLAUSES* EMB * 4);
    float* C_c   = (float*)bump(wp, (size_t)NCLAUSES* EMB * 4);
    float* LN_L  = (float*)bump(wp, (size_t)NLITS   * EMB * 4);
    float* T1    = (float*)bump(wp, (size_t)NLITS   * EMB * 4);
    float* T2    = (float*)bump(wp, (size_t)NLITS   * EMB * 4);
    float* Lpre  = (float*)bump(wp, (size_t)NLITS   * EMB * 4);
    float* LN_C  = (float*)bump(wp, (size_t)NCLAUSES* EMB * 4);
    float* Tc1   = (float*)bump(wp, (size_t)NCLAUSES* EMB * 4);
    float* Tc2   = (float*)bump(wp, (size_t)NCLAUSES* EMB * 4);
    float* Cpre  = (float*)bump(wp, (size_t)NCLAUSES* EMB * 4);
    float* LC    = (float*)bump(wp, (size_t)NCLAUSES* EMB * 4);
    float* CL    = (float*)bump(wp, (size_t)NLITS   * EMB * 4);
    float* gatesC= (float*)bump(wp, (size_t)NCLAUSES* 4 * EMB * 4);
    float* gatesL= (float*)bump(wp, (size_t)NLITS   * 4 * EMB * 4);
    float* voteBuf=(float*)bump(wp, (size_t)NLITS * 4);
    float* biasC = (float*)bump(wp, 512 * 4);
    float* biasL = (float*)bump(wp, 512 * 4);
    _Float16* pLm1 = (_Float16*)bump(wp, 128 * 128 * 2);
    _Float16* pLm2 = (_Float16*)bump(wp, 128 * 128 * 2);
    _Float16* pLm3 = (_Float16*)bump(wp, 128 * 128 * 2);
    _Float16* pCm1 = (_Float16*)bump(wp, 128 * 128 * 2);
    _Float16* pCm2 = (_Float16*)bump(wp, 128 * 128 * 2);
    _Float16* pCm3 = (_Float16*)bump(wp, 128 * 128 * 2);
    _Float16* pV1  = (_Float16*)bump(wp, 128 * 128 * 2);
    _Float16* pV2  = (_Float16*)bump(wp, 128 * 128 * 2);
    _Float16* pWihX1 = (_Float16*)bump(wp, 128 * 512 * 2);   // Lu_wih cols [0:128)
    _Float16* pWihX2 = (_Float16*)bump(wp, 128 * 512 * 2);   // Lu_wih cols [128:256)
    _Float16* pWhhL  = (_Float16*)bump(wp, 128 * 512 * 2);
    _Float16* pWihC  = (_Float16*)bump(wp, 128 * 512 * 2);
    _Float16* pWhhC  = (_Float16*)bump(wp, 128 * 512 * 2);

    float* out       = (float*)d_out;
    float* out_votes = out + 16;                       // [8,16,128,2]
    float* out_Ls    = out + 16 + 32768;               // [8,4096,128]
    float* out_Cs    = out + 16 + 32768 + 8 * NLITS * EMB;  // [8,8192,128]

    // ---- one-time prep: pack weights to f16 B-fragments, combine biases --
    auto packGrid = [](int n) { return dim3((n + 255) / 256); };
    pack_b_kernel<<<packGrid(128*128), 256, 0, stream>>>(Lm_w1, pLm1, 128, 128, 128, 0);
    pack_b_kernel<<<packGrid(128*128), 256, 0, stream>>>(Lm_w2, pLm2, 128, 128, 128, 0);
    pack_b_kernel<<<packGrid(128*128), 256, 0, stream>>>(Lm_w3, pLm3, 128, 128, 128, 0);
    pack_b_kernel<<<packGrid(128*128), 256, 0, stream>>>(Cm_w1, pCm1, 128, 128, 128, 0);
    pack_b_kernel<<<packGrid(128*128), 256, 0, stream>>>(Cm_w2, pCm2, 128, 128, 128, 0);
    pack_b_kernel<<<packGrid(128*128), 256, 0, stream>>>(Cm_w3, pCm3, 128, 128, 128, 0);
    pack_b_kernel<<<packGrid(128*128), 256, 0, stream>>>(Lv_w1, pV1,  128, 128, 128, 0);
    pack_b_kernel<<<packGrid(128*128), 256, 0, stream>>>(Lv_w2, pV2,  128, 128, 128, 0);
    pack_b_kernel<<<packGrid(512*128), 256, 0, stream>>>(Lu_wih, pWihX1, 512, 128, 256, 0);
    pack_b_kernel<<<packGrid(512*128), 256, 0, stream>>>(Lu_wih, pWihX2, 512, 128, 256, 128);
    pack_b_kernel<<<packGrid(512*128), 256, 0, stream>>>(Lu_whh, pWhhL,  512, 128, 128, 0);
    pack_b_kernel<<<packGrid(512*128), 256, 0, stream>>>(Cu_wih, pWihC,  512, 128, 128, 0);
    pack_b_kernel<<<packGrid(512*128), 256, 0, stream>>>(Cu_whh, pWhhC,  512, 128, 128, 0);
    add_vec_kernel<<<2, 256, 0, stream>>>(Cu_bih, Cu_bhh, biasC, 512);
    add_vec_kernel<<<2, 256, 0, stream>>>(Lu_bih, Lu_bhh, biasL, 512);

    // ---- init hidden states ---------------------------------------------
    init_state_kernel<<<(NLITS*EMB+255)/256, 256, 0, stream>>>(L_h, LiW, LiB, NLITS);
    init_state_kernel<<<(NCLAUSES*EMB+255)/256, 256, 0, stream>>>(C_h, CiW, CiB, NCLAUSES);
    zero_kernel<<<(NLITS*EMB+255)/256, 256, 0, stream>>>(L_c, NLITS*EMB);
    zero_kernel<<<(NCLAUSES*EMB+255)/256, 256, 0, stream>>>(C_c, NCLAUSES*EMB);

    auto gemm = [&](const float* A, const _Float16* B, const float* bias, float* Cd,
                    int M, int N, int K, int rowXor, int acc, int relu) {
        dim3 grid(M / 128, N / (16 * NT));
        gemm_wmma_kernel<<<grid, 256, 0, stream>>>(A, B, bias, Cd, M, N, K, rowXor, acc, relu);
    };

    for (int it = 0; it < N_ITERS; ++it) {
        // L message MLP on LN(L_h)
        layernorm_kernel<<<NLITS/8, 256, 0, stream>>>(L_h, ln_l_g, ln_l_b, LN_L, NLITS);
        gemm(LN_L, pLm1, Lm_b1, T1,   NLITS, 128, 128, 0, 0, 1);
        gemm(T1,   pLm2, Lm_b2, T2,   NLITS, 128, 128, 0, 0, 1);
        gemm(T2,   pLm3, Lm_b3, Lpre, NLITS, 128, 128, 0, 0, 0);

        // LC = conn^T @ Lpre  (sparse scatter over edges)
        zero_kernel<<<(NCLAUSES*EMB+255)/256, 256, 0, stream>>>(LC, NCLAUSES*EMB);
        scatter_add_kernel<<<(NCELLS*EMB+255)/256, 256, 0, stream>>>(e_lit, e_cls, ew, Lpre, LC);

        // C-LSTM: gates = LC@Wih^T + C_h@Whh^T + (bih+bhh)
        gemm(LC,  pWihC, biasC,   gatesC, NCLAUSES, 512, 128, 0, 0, 0);
        gemm(C_h, pWhhC, nullptr, gatesC, NCLAUSES, 512, 128, 0, 1, 0);
        lstm_cell_kernel<<<(NCLAUSES*EMB+255)/256, 256, 0, stream>>>(C_h, C_c, gatesC, NCLAUSES);

        // C message MLP on LN(C_h)
        layernorm_kernel<<<NCLAUSES/8, 256, 0, stream>>>(C_h, ln_c_g, ln_c_b, LN_C, NCLAUSES);
        gemm(LN_C, pCm1, Cm_b1, Tc1,  NCLAUSES, 128, 128, 0, 0, 1);
        gemm(Tc1,  pCm2, Cm_b2, Tc2,  NCLAUSES, 128, 128, 0, 0, 1);
        gemm(Tc2,  pCm3, Cm_b3, Cpre, NCLAUSES, 128, 128, 0, 0, 0);

        // CL = conn @ Cpre
        zero_kernel<<<(NLITS*EMB+255)/256, 256, 0, stream>>>(CL, NLITS*EMB);
        scatter_add_kernel<<<(NCELLS*EMB+255)/256, 256, 0, stream>>>(e_cls, e_lit, ew, Cpre, CL);

        // L-LSTM: x = [CL, flipped(L_h)]; flip == row XOR 2048
        gemm(CL,  pWihX1, biasL,   gatesL, NLITS, 512, 128, 0,    0, 0);
        gemm(L_h, pWihX2, nullptr, gatesL, NLITS, 512, 128, 2048, 1, 0);
        gemm(L_h, pWhhL,  nullptr, gatesL, NLITS, 512, 128, 0,    1, 0);
        lstm_cell_kernel<<<(NLITS*EMB+255)/256, 256, 0, stream>>>(L_h, L_c, gatesL, NLITS);

        // votes
        gemm(L_h, pV1, Lv_b1, T1, NLITS, 128, 128, 0, 0, 1);
        gemm(T1,  pV2, Lv_b2, T2, NLITS, 128, 128, 0, 0, 1);
        vote_finish_kernel<<<NLITS/8, 256, 0, stream>>>(T2, Lv_w3, Lv_b3, out_votes, voteBuf, it);

        // record Ls / Cs
        (void)hipMemcpyAsync(out_Ls + (size_t)it * NLITS * EMB, L_h,
                             (size_t)NLITS * EMB * 4, hipMemcpyDeviceToDevice, stream);
        (void)hipMemcpyAsync(out_Cs + (size_t)it * NCLAUSES * EMB, C_h,
                             (size_t)NCLAUSES * EMB * 4, hipMemcpyDeviceToDevice, stream);
    }

    vote_mean_kernel<<<NPROB, 128, 0, stream>>>(voteBuf, out);
}
